// MultiHeadAttention_53437983097248
// MI455X (gfx1250) — matrix operand order repaired and checked by reading, not compiled, not run on hardware
//
#include <hip/hip_runtime.h>
#include <hip/hip_bf16.h>

// ---------------------------------------------------------------------------
// MultiHeadAttention forward for MI455X (gfx1250), bf16 WMMA pipeline.
// Shapes: B=4, S=2048, H=16, Dk=Dv=64, Dmodel=1024.
// Outputs (concatenated): ln [B,S,1024] f32, attn [B,H,S,S] f32.
// ---------------------------------------------------------------------------

#define BB 4
#define SS 2048
#define HH 16
#define DD 64
#define DM 1024
#define MM (BB * SS) // 8192 token rows
#define QT 32        // q rows per attention workgroup

#define SCP (SS + 4)  // padded LDS stride for score rows (16B-aligned, bank-skewed)
#define CXP (DD + 1)  // padded stride for ctx accumulator
#define OTP (DM + 4)  // padded stride for output tile

typedef __attribute__((ext_vector_type(16))) __bf16 v16bf;
typedef __attribute__((ext_vector_type(8)))  __bf16 v8bf;
typedef __attribute__((ext_vector_type(8)))  float  v8f;

__device__ __forceinline__ v8f wmma_bf16(v16bf a, v16bf b, v8f c) {
  return __builtin_amdgcn_wmma_f32_16x16x32_bf16(
      /*neg_a=*/false, a, /*neg_b=*/false, b,
      /*c_mod=*/(short)0, c, /*reuse_a=*/false, /*reuse_b=*/false);
}

// A-matrix fragment (16x32 bf16, row-major source, this lane's row pointer).
// Lane rows = L%16; elements 0..15 hold k = (L/16)*16 + 0..15 (matches B).
__device__ __forceinline__ v16bf a_frag(const __bf16* rowptr, int k0, int lane) {
  const int hh = lane >> 4;
  v8bf lo = *(const v8bf*)(rowptr + k0 + hh * 16);
  v8bf hi = *(const v8bf*)(rowptr + k0 + hh * 16 + 8);
  v16bf a;
#pragma unroll
  for (int i = 0; i < 8; ++i) { a[i] = lo[i]; a[i + 8] = hi[i]; }
  return a;
}

// ---------------------------------------------------------------------------
// Kernel 1: f32 -> bf16 conversion (weights + activations).
// ---------------------------------------------------------------------------
__global__ void k_f32_to_bf16(const float* __restrict__ src,
                              __bf16* __restrict__ dst, int n) {
  int i = blockIdx.x * blockDim.x + threadIdx.x;
  int stride = gridDim.x * blockDim.x;
  for (; i < n; i += stride) dst[i] = (__bf16)src[i];
}

// ---------------------------------------------------------------------------
// Kernel 2: projection GEMM  Y[m,n] = sum_k X[m,k] * W[n,k]
// X: [MM, DM] bf16 row-major, W: [DM, DM] bf16 row-major ([e_out][d_in]).
// mode 0: write head-major [BH, S, 64]; mode 1: write transposed [BH, 64, S]
//          (staged through LDS so the transposed store is coalesced).
// Workgroup tile 64x64, 8 waves, each wave two 16x16 subtiles sharing A.
// ---------------------------------------------------------------------------
__global__ void __launch_bounds__(256)
k_proj(const __bf16* __restrict__ X, const __bf16* __restrict__ W,
       __bf16* __restrict__ Out, int mode) {
  __shared__ __bf16 vtile[64][72]; // 9216B, used only in mode 1

  const int tid = threadIdx.x;
  const int lane = tid & 31;
  const int wv = tid >> 5;
  const int nblk = blockIdx.x & 15;   // 16 column blocks of 64 (== head id)
  const int mblk = blockIdx.x >> 4;   // 128 row blocks of 64
  const int mi = wv >> 1;             // 0..3 : 16-row subtile
  const int n0 = nblk * 64 + (wv & 1) * 32; // first of two n-tiles
  const int r = lane & 15, hh = lane >> 4;

  const __bf16* arow  = X + (size_t)(mblk * 64 + mi * 16 + r) * DM;
  const __bf16* brow0 = W + (size_t)(n0 + r) * DM;        // B col = lane&15
  const __bf16* brow1 = brow0 + (size_t)16 * DM;

  v8f c0 = {}, c1 = {};
  for (int k0 = 0; k0 < DM; k0 += 32) {
    __builtin_prefetch(arow + k0 + 256, 0, 3); // global_prefetch_b8
    v16bf a  = a_frag(arow, k0, lane);
    v16bf b0 = *(const v16bf*)(brow0 + k0 + hh * 16);
    v16bf b1 = *(const v16bf*)(brow1 + k0 + hh * 16);
    c0 = wmma_bf16(a, b0, c0);
    c1 = wmma_bf16(a, b1, c1);
  }

  const int m0 = mblk * 64 + mi * 16;
  if (mode == 0) {
#pragma unroll
    for (int t = 0; t < 2; ++t) {
      v8f cz = t ? c1 : c0;
      int n = n0 + t * 16 + r;          // D col = lane&15
      int head = n >> 6, d = n & 63;
#pragma unroll
      for (int j = 0; j < 8; ++j) {
        int m = m0 + hh * 8 + j;        // D rows = 8*(lane>>4)+j
        int bb = m >> 11, s = m & (SS - 1);
        Out[(size_t)((bb * HH + head) * SS + s) * DD + d] = (__bf16)cz[j];
      }
    }
  } else {
    // Stage transposed tile in LDS, then store coalesced along s.
#pragma unroll
    for (int t = 0; t < 2; ++t) {
      v8f cz = t ? c1 : c0;
      int nl = (wv & 1) * 32 + t * 16 + r;   // local d
#pragma unroll
      for (int j = 0; j < 8; ++j) {
        int ml = mi * 16 + hh * 8 + j;       // local s
        vtile[nl][ml] = (__bf16)cz[j];
      }
    }
    __syncthreads();
    const int bb = (mblk * 64) >> 11;
    const int s0 = (mblk * 64) & (SS - 1);
    const int head = nblk;                   // 64-wide n-block == one head
    const int d = tid >> 2;                  // 0..63
    const int sseg = (tid & 3) * 16;         // 16 bf16 = 32B per thread
    v16bf v = *(const v16bf*)&vtile[d][sseg];
    *(v16bf*)(Out + (size_t)((bb * HH + head) * DD + d) * SS + s0 + sseg) = v;
  }
}

// ---------------------------------------------------------------------------
// Kernel 3: fused attention per (b,h, 32-row q tile).
//   scores (WMMA) -> LDS f32 [32][SS] (padded stride, ~263KB of 320KB WGP LDS)
//   masked softmax (LDS tree reductions) -> write attn to d_out
//   context = P @ V  (WMMA, K split across 8 waves, ds_add_f32 reduction)
// ---------------------------------------------------------------------------
__global__ void __launch_bounds__(256)
k_attn(const __bf16* __restrict__ Qh, const __bf16* __restrict__ Kh,
       const __bf16* __restrict__ Vt, const unsigned char* __restrict__ mask,
       float* __restrict__ attn_out, __bf16* __restrict__ ctx) {
  __shared__ float sc[QT][SCP];      // 262656 B
  __shared__ float ctxacc[QT][CXP];  // 8320 B
  __shared__ float part[QT][8];
  __shared__ float rowm[QT];
  __shared__ float rowinv[QT];

  const int tid = threadIdx.x, lane = tid & 31, wv = tid >> 5;
  const int qblk = blockIdx.x & (SS / QT - 1); // 64 q tiles
  const int bh = blockIdx.x / (SS / QT);       // 0..63
  const int bidx = bh >> 4;                    // batch
  const int head = bh & 15;
  const int q0 = qblk * QT;
  const int r = lane & 15, hh = lane >> 4;

  const __bf16* Qb = Qh + (size_t)bh * SS * DD;
  const __bf16* Kb = Kh + (size_t)bh * SS * DD;
  const __bf16* Vb = Vt + (size_t)bh * DD * SS;
  const unsigned char* mb = mask + (size_t)bidx * SS * SS;
  float* attn_b = attn_out + (size_t)bh * SS * SS;

  for (int i = tid; i < QT * CXP; i += 256) ((float*)ctxacc)[i] = 0.f;

  // ---- Phase 1: scores = (Q K^T) / 8, masked, into LDS ----
  v16bf aq[2][2];
#pragma unroll
  for (int m = 0; m < 2; ++m) {
    const __bf16* arow = Qb + (size_t)(q0 + m * 16 + r) * DD;
    aq[m][0] = a_frag(arow, 0, lane);
    aq[m][1] = a_frag(arow, 32, lane);
  }
  const float scl = 0.125f; // 1/sqrt(64)
  for (int nt = 0; nt < 16; ++nt) {                 // 16 key tiles per wave
    const int t0 = (wv * 16 + nt) * 16;
    const __bf16* brow = Kb + (size_t)(t0 + r) * DD;
    __builtin_prefetch(brow + 16 * DD, 0, 3);
    v16bf b0 = *(const v16bf*)(brow + hh * 16);
    v16bf b1 = *(const v16bf*)(brow + 32 + hh * 16);
#pragma unroll
    for (int m = 0; m < 2; ++m) {
      v8f c = {};
      c = wmma_bf16(aq[m][0], b0, c);
      c = wmma_bf16(aq[m][1], b1, c);
      const int t = t0 + r;
#pragma unroll
      for (int j = 0; j < 8; ++j) {
        int ql = m * 16 + hh * 8 + j;
        float v = c[j] * scl;
        if (mb[(size_t)(q0 + ql) * SS + t]) v = -1e9f;
        sc[ql][t] = v;
      }
    }
  }
  __syncthreads();

  // ---- Phase 2: softmax over each row of 2048 (interleaved, 8 threads per row) ----
  const int qq = tid >> 3, l8 = tid & 7;
  {
    const float* rp = sc[qq];
    float mx = -3.0e38f;
    for (int i = 0; i < 256; ++i) mx = fmaxf(mx, rp[l8 + 8 * i]);
    part[qq][l8] = mx;
  }
  __syncthreads();
  if (tid < QT) {
    float mx = part[tid][0];
    for (int i = 1; i < 8; ++i) mx = fmaxf(mx, part[tid][i]);
    rowm[tid] = mx;
  }
  __syncthreads();
  {
    float* rp = sc[qq];
    float mx = rowm[qq], s = 0.f;
    for (int i = 0; i < 256; ++i) {
      float e = __expf(rp[l8 + 8 * i] - mx);
      rp[l8 + 8 * i] = e;
      s += e;
    }
    part[qq][l8] = s;
  }
  __syncthreads();
  if (tid < QT) {
    float s = 0.f;
    for (int i = 0; i < 8; ++i) s += part[tid][i];
    rowinv[tid] = 1.0f / s;
  }
  __syncthreads();
  for (int q = 0; q < QT; ++q) {     // normalize + coalesced attn write
    float ri = rowinv[q];
    float* rp = sc[q];
    float* gp = attn_b + (size_t)(q0 + q) * SS;
    for (int t = tid; t < SS; t += 256) {
      float p = rp[t] * ri;
      rp[t] = p;
      gp[t] = p;
    }
  }
  __syncthreads();

  // ---- Phase 3: context = P @ V, K-dim split across 8 waves ----
  v8f acc[2][4] = {};
  const int kbase = wv * 256;
  for (int kc = 0; kc < 8; ++kc) {
    const int k0 = kbase + kc * 32;
    v16bf a[2];
#pragma unroll
    for (int m = 0; m < 2; ++m) {
      const float* rp = sc[m * 16 + r] + k0;
      v16bf t;
#pragma unroll
      for (int i = 0; i < 8; ++i) {
        t[i]     = (__bf16)rp[hh * 16 + i];
        t[i + 8] = (__bf16)rp[hh * 16 + 8 + i];
      }
      a[m] = t;
    }
#pragma unroll
    for (int nt = 0; nt < 4; ++nt) {
      const __bf16* brow = Vb + (size_t)(nt * 16 + r) * SS; // V^T row = d
      __builtin_prefetch(brow + k0 + 32, 0, 3);
      v16bf bf = *(const v16bf*)(brow + k0 + hh * 16);
      acc[0][nt] = wmma_bf16(a[0], bf, acc[0][nt]);
      acc[1][nt] = wmma_bf16(a[1], bf, acc[1][nt]);
    }
  }
#pragma unroll
  for (int m = 0; m < 2; ++m)
#pragma unroll
    for (int nt = 0; nt < 4; ++nt)
#pragma unroll
      for (int j = 0; j < 8; ++j)
        atomicAdd(&ctxacc[m * 16 + hh * 8 + j][nt * 16 + r], acc[m][nt][j]);
  __syncthreads();

  for (int i = tid; i < QT * DD; i += 256) {
    int q = i >> 6, d = i & 63;
    ctx[((size_t)(bidx * SS) + q0 + q) * DM + head * 64 + d] =
        (__bf16)ctxacc[q][d];
  }
}

// ---------------------------------------------------------------------------
// Kernel 4: out = ctx @ Wfc^T + residual, then LayerNorm. 16-row tiles.
// ---------------------------------------------------------------------------
__global__ void __launch_bounds__(256)
k_out_ln(const __bf16* __restrict__ ctx, const __bf16* __restrict__ Wfc,
         const float* __restrict__ resid, float* __restrict__ ln_out) {
  __shared__ float outt[16][OTP]; // ~64KB
  __shared__ float ps[16][16];
  __shared__ float mu[16];
  __shared__ float sig[16];

  const int tid = threadIdx.x, lane = tid & 31, wv = tid >> 5;
  const int m0 = blockIdx.x * 16;
  const int r = lane & 15, hh = lane >> 4;
  const int nb = wv * 128; // each wave owns 8 n-tiles

  const __bf16* arow = ctx + (size_t)(m0 + r) * DM;
  v8f acc[8] = {};
  for (int k0 = 0; k0 < DM; k0 += 32) {
    __builtin_prefetch(arow + k0 + 256, 0, 3);
    v16bf a = a_frag(arow, k0, lane);
#pragma unroll
    for (int nt = 0; nt < 8; ++nt) {
      const __bf16* brow = Wfc + (size_t)(nb + nt * 16 + r) * DM;
      v16bf b = *(const v16bf*)(brow + k0 + hh * 16);
      acc[nt] = wmma_bf16(a, b, acc[nt]);
    }
  }
#pragma unroll
  for (int nt = 0; nt < 8; ++nt)
#pragma unroll
    for (int j = 0; j < 8; ++j)
      outt[hh * 8 + j][nb + nt * 16 + r] = acc[nt][j];
  __syncthreads();

  // residual + LayerNorm; per-thread columns interleaved: d = l16 + 16*i
  const int rr = tid >> 4, l16 = tid & 15;
  {
    float s = 0.f;
    float* op = outt[rr];
    const float* ip = resid + (size_t)(m0 + rr) * DM;
    for (int i = 0; i < 64; ++i) {
      int d = l16 + 16 * i;
      float x = op[d] + ip[d];
      op[d] = x;
      s += x;
    }
    ps[rr][l16] = s;
  }
  __syncthreads();
  if (tid < 16) {
    float s = 0.f;
    for (int i = 0; i < 16; ++i) s += ps[tid][i];
    mu[tid] = s * (1.0f / DM);
  }
  __syncthreads();
  {
    float m = mu[rr], s = 0.f;
    const float* op = outt[rr];
    for (int i = 0; i < 64; ++i) {
      float d = op[l16 + 16 * i] - m;
      s += d * d;
    }
    ps[rr][l16] = s;
  }
  __syncthreads();
  if (tid < 16) {
    float s = 0.f;
    for (int i = 0; i < 16; ++i) s += ps[tid][i];
    sig[tid] = rsqrtf(s * (1.0f / DM) + 1e-5f);
  }
  __syncthreads();
  {
    float m = mu[rr], rs = sig[rr];
    const float* op = outt[rr];
    float* gp = ln_out + (size_t)(m0 + rr) * DM;
    for (int i = 0; i < 64; ++i) {
      int d = l16 + 16 * i;
      gp[d] = (op[d] - m) * rs;
    }
  }
}

// ---------------------------------------------------------------------------
extern "C" void kernel_launch(void* const* d_in, const int* in_sizes, int n_in,
                              void* d_out, int out_size, void* d_ws,
                              size_t ws_size, hipStream_t stream) {
  (void)in_sizes; (void)n_in; (void)out_size; (void)ws_size;
  const float* inQ  = (const float*)d_in[0];
  const float* inK  = (const float*)d_in[1];
  const float* inV  = (const float*)d_in[2];
  const unsigned char* mask = (const unsigned char*)d_in[3]; // bool mask
  const float* Wq   = (const float*)d_in[4];
  const float* Wk   = (const float*)d_in[5];
  const float* Wv   = (const float*)d_in[6];
  const float* Wfc  = (const float*)d_in[7];

  float* ln_out   = (float*)d_out;
  float* attn_out = ln_out + (size_t)BB * SS * DM;

  // Workspace layout (bf16 elements)
  __bf16* p = (__bf16*)d_ws;
  const size_t NW = (size_t)DM * DM;  // 1M
  const size_t NX = (size_t)MM * DM;  // 8M
  __bf16* Wq_b = p; p += NW;
  __bf16* Wk_b = p; p += NW;
  __bf16* Wv_b = p; p += NW;
  __bf16* Wfc_b = p; p += NW;
  __bf16* Xq_b = p; p += NX;
  __bf16* Xk_b = p; p += NX;
  __bf16* Xv_b = p; p += NX;
  __bf16* Qh = p; p += NX;   // [BH,S,64]
  __bf16* Kh = p; p += NX;   // [BH,S,64]
  __bf16* Vt = p; p += NX;   // [BH,64,S]
  __bf16* ctxb = p; p += NX; // [B*S, 1024]

  // 1) conversions
  k_f32_to_bf16<<<1024, 256, 0, stream>>>(Wq,  Wq_b,  (int)NW);
  k_f32_to_bf16<<<1024, 256, 0, stream>>>(Wk,  Wk_b,  (int)NW);
  k_f32_to_bf16<<<1024, 256, 0, stream>>>(Wv,  Wv_b,  (int)NW);
  k_f32_to_bf16<<<1024, 256, 0, stream>>>(Wfc, Wfc_b, (int)NW);
  k_f32_to_bf16<<<2048, 256, 0, stream>>>(inQ, Xq_b,  (int)NX);
  k_f32_to_bf16<<<2048, 256, 0, stream>>>(inK, Xk_b,  (int)NX);
  k_f32_to_bf16<<<2048, 256, 0, stream>>>(inV, Xv_b,  (int)NX);

  // 2) projections (grid = 128 m-blocks x 16 n-blocks)
  k_proj<<<2048, 256, 0, stream>>>(Xq_b, Wq_b, Qh, 0);
  k_proj<<<2048, 256, 0, stream>>>(Xk_b, Wk_b, Kh, 0);
  k_proj<<<2048, 256, 0, stream>>>(Xv_b, Wv_b, Vt, 1);

  // 3) fused attention: B*H * (S/QT) workgroups
  k_attn<<<BB * HH * (SS / QT), 256, 0, stream>>>(Qh, Kh, Vt, mask, attn_out,
                                                  ctxb);

  // 4) output projection + residual + LayerNorm
  k_out_ln<<<MM / 16, 256, 0, stream>>>(ctxb, Wfc_b, inQ, ln_out);
}
